// RelationNet_72919954752142
// MI455X (gfx1250) — matrix operand order repaired and checked by reading, not compiled, not run on hardware
//
#include <hip/hip_runtime.h>
#include <hip/hip_bf16.h>

// ---------------- vector types ----------------
typedef __attribute__((ext_vector_type(8)))  __bf16 v8bf;
typedef __attribute__((ext_vector_type(16))) __bf16 v16bf;
typedef __attribute__((ext_vector_type(8)))  float  v8f;

__device__ __forceinline__ __bf16 f2bf(float f) {
    union { float f; unsigned u; } v; v.f = f;
    unsigned r = v.u + 0x7FFFu + ((v.u >> 16) & 1u);   // round-to-nearest-even
    unsigned short h = (unsigned short)(r >> 16);
    return __builtin_bit_cast(__bf16, h);
}

// ---------------- problem constants ----------------
#define NB   32      // batch
#define C1   24      // stem channels
#define H1   32      // after conv1
#define H2   16      // after conv2
#define HW2  256     // 16*16
#define DREL 180     // pair feature dim
#define KRELP 192    // DREL padded to multiple of 32
#define MREL 128     // relation out dim
#define PROJ 512
#define FCD  1024
#define NA   2
#define PAIRS 32640.0f   // 256*255/2

// ======================================================================
// conv1: image[32,3,64,64] -> y1[32,24,32,32] (conv3x3 s2 p1 + bias, raw)
// ======================================================================
__global__ void conv1_kernel(const float* __restrict__ img,
                             const float* __restrict__ W,
                             const float* __restrict__ b,
                             float* __restrict__ y1) {
    int idx = blockIdx.x * blockDim.x + threadIdx.x;
    const int total = NB * C1 * H1 * H1;
    if (idx >= total) return;
    int ow = idx & 31; int t = idx >> 5;
    int oh = t & 31;   t >>= 5;
    int c  = t % C1;   int n = t / C1;
    float acc = b[c];
    #pragma unroll
    for (int ic = 0; ic < 3; ++ic) {
        #pragma unroll
        for (int kh = 0; kh < 3; ++kh) {
            int ih = oh * 2 - 1 + kh;
            if (ih < 0 || ih >= 64) continue;
            #pragma unroll
            for (int kw = 0; kw < 3; ++kw) {
                int iw = ow * 2 - 1 + kw;
                if (iw < 0 || iw >= 64) continue;
                acc += img[((n * 3 + ic) * 64 + ih) * 64 + iw] *
                       W[((c * 3 + ic) * 3 + kh) * 3 + kw];
            }
        }
    }
    y1[idx] = acc;
}

// ======================================================================
// BN batch statistics per channel: mean, rstd   (one block per channel)
// ======================================================================
__global__ void bn_stats_kernel(const float* __restrict__ y,
                                float* __restrict__ mean,
                                float* __restrict__ rstd,
                                int HW) {
    int c = blockIdx.x;
    int cnt = NB * HW;
    float s = 0.f, sq = 0.f;
    for (int t = threadIdx.x; t < cnt; t += blockDim.x) {
        int n = t / HW, hw = t - n * HW;
        float v = y[(n * C1 + c) * HW + hw];
        s += v; sq += v * v;
    }
    __shared__ float shs[256], shq[256];
    shs[threadIdx.x] = s; shq[threadIdx.x] = sq;
    __syncthreads();
    for (int off = 128; off > 0; off >>= 1) {
        if (threadIdx.x < off) {
            shs[threadIdx.x] += shs[threadIdx.x + off];
            shq[threadIdx.x] += shq[threadIdx.x + off];
        }
        __syncthreads();
    }
    if (threadIdx.x == 0) {
        float mu  = shs[0] / (float)cnt;
        float var = shq[0] / (float)cnt - mu * mu;
        mean[c] = mu;
        rstd[c] = rsqrtf(var + 1e-5f);
    }
}

// ======================================================================
// conv2: reads y1 with BN1+ReLU fused on the fly -> y2[32,24,16,16] raw
// ======================================================================
__global__ void conv2_kernel(const float* __restrict__ y1,
                             const float* __restrict__ mu1,
                             const float* __restrict__ rs1,
                             const float* __restrict__ g1,
                             const float* __restrict__ be1,
                             const float* __restrict__ W,
                             const float* __restrict__ b,
                             float* __restrict__ y2) {
    int idx = blockIdx.x * blockDim.x + threadIdx.x;
    const int total = NB * C1 * H2 * H2;
    if (idx >= total) return;
    int ow = idx & 15; int t = idx >> 4;
    int oh = t & 15;   t >>= 4;
    int c  = t % C1;   int n = t / C1;
    float acc = b[c];
    for (int ic = 0; ic < C1; ++ic) {
        float sc = rs1[ic] * g1[ic];
        float sh = be1[ic] - mu1[ic] * sc;
        #pragma unroll
        for (int kh = 0; kh < 3; ++kh) {
            int ih = oh * 2 - 1 + kh;
            if (ih < 0 || ih >= H1) continue;
            #pragma unroll
            for (int kw = 0; kw < 3; ++kw) {
                int iw = ow * 2 - 1 + kw;
                if (iw < 0 || iw >= H1) continue;
                float x = y1[((n * C1 + ic) * H1 + ih) * H1 + iw] * sc + sh;
                x = fmaxf(x, 0.f);
                acc += x * W[((c * C1 + ic) * 3 + kh) * 3 + kw];
            }
        }
    }
    y2[idx] = acc;
}

// ======================================================================
// Pack f32 weight matrix B[K,N] into per-lane bf16 WMMA fragment layout,
// K zero-padded to ksteps*32.  Element index:
//   Bp[ ((ntile*ksteps + kstep)*32 + lane)*16 + e ],
//   k = kstep*32 + (lane>>4)*16 + e,  n = ntile*16 + (lane&15)
// ======================================================================
__global__ void pack_B_kernel(const float* __restrict__ B,
                              __bf16* __restrict__ Bp,
                              int N, int K, int ksteps) {
    int idx = blockIdx.x * blockDim.x + threadIdx.x;
    int total = (N / 16) * ksteps * 32 * 16;
    if (idx >= total) return;
    int e = idx & 15; int t = idx >> 4;
    int lane = t & 31; t >>= 5;
    int kstep = t % ksteps; int ntile = t / ksteps;
    int k = kstep * 32 + (lane >> 4) * 16 + e;
    int n = ntile * 16 + (lane & 15);
    Bp[idx] = (k < K) ? f2bf(B[k * N + n]) : f2bf(0.f);
}

// ======================================================================
// Build relation-GEMM A matrix directly in bf16, K padded to 192:
//   row n = [ sum_i (255-i)*fc[n,i] | sum_i i*fc[n,i] | P*question[n] | 0pad ]
// fc[n,i] = [ relu(bn2(y2[n,:,i])) , i%16 , i/16 ]
// One block per n, 256 threads = one per spatial position i.
// ======================================================================
__global__ void build_relA_kernel(const float* __restrict__ y2,
                                  const float* __restrict__ mu2,
                                  const float* __restrict__ rs2,
                                  const float* __restrict__ g2,
                                  const float* __restrict__ be2,
                                  const float* __restrict__ question,
                                  __bf16* __restrict__ Arel) {
    int n = blockIdx.x;
    int i = threadIdx.x;                // position 0..255
    __shared__ float s1[26], s2[26];
    if (i < 26) { s1[i] = 0.f; s2[i] = 0.f; }
    __syncthreads();
    float w1 = (float)(255 - i);
    float w2 = (float)i;
    for (int c = 0; c < C1; ++c) {
        float sc = rs2[c] * g2[c];
        float v  = (y2[(n * C1 + c) * HW2 + i] - mu2[c]) * sc + be2[c];
        v = fmaxf(v, 0.f);
        atomicAdd(&s1[c], w1 * v);      // ds_add_f32
        atomicAdd(&s2[c], w2 * v);
    }
    float cx = (float)(i & 15);
    float cy = (float)(i >> 4);
    atomicAdd(&s1[24], w1 * cx); atomicAdd(&s1[25], w1 * cy);
    atomicAdd(&s2[24], w2 * cx); atomicAdd(&s2[25], w2 * cy);
    __syncthreads();
    __bf16* row = Arel + n * KRELP;
    if (i < 26) { row[i] = f2bf(s1[i]); row[26 + i] = f2bf(s2[i]); }
    if (i < 128) row[52 + i] = f2bf(PAIRS * question[n * 128 + i]);
    if (i < KRELP - DREL) row[DREL + i] = f2bf(0.f);    // zero the K pad
}

// ======================================================================
// Wave32 WMMA GEMM:  C = act( A @ B + bias_scale*bias )
//   A:  bf16 row-major [M x ksteps*32]  (pre-padded with zeros)
//   Bp: bf16 packed per-lane fragments  (from pack_B_kernel)
//   Outputs both f32 C and bf16 C (the bf16 copy feeds the next GEMM's A).
// One 16x16 output tile per wave (block = 32 threads).
// Inner loop: 4x global_load_b128 + 1x v_wmma_f32_16x16x32_bf16.
// ======================================================================
__global__ void gemm_bf16_wmma_kernel(const __bf16* __restrict__ A,
                                      const __bf16* __restrict__ Bp,
                                      const float* __restrict__ bias,
                                      float* __restrict__ Cf,
                                      __bf16* __restrict__ Cbf,
                                      int N, int ksteps,
                                      float bias_scale, int relu) {
    const int lane = threadIdx.x;       // 0..31
    const int half = lane >> 4;         // lane-half selects K sub-range
    const int l    = lane & 15;
    const int mtile = blockIdx.y;
    const int ntile = blockIdx.x;
    const int Kpad  = ksteps * 32;
    const int rowA  = mtile * 16 + l;
    const int coln  = ntile * 16 + l;

    const __bf16* ap = A + (size_t)rowA * Kpad + half * 16;  // 16B aligned
    const __bf16* bp = Bp + ((size_t)ntile * ksteps) * 512 + lane * 16; // 32B aligned

    v8f acc = {};
    for (int ks = 0; ks < ksteps; ++ks) {
        v8bf alo = *(const v8bf*)(ap);          // K = k0 + half*16 + 0..7
        v8bf ahi = *(const v8bf*)(ap + 8);      // K = k0 + half*16 + 8..15
        v16bf a = __builtin_shufflevector(alo, ahi,
                    0, 1, 2, 3, 4, 5, 6, 7, 8, 9, 10, 11, 12, 13, 14, 15);
        v16bf b = *(const v16bf*)(bp);          // contiguous packed fragment
        acc = __builtin_amdgcn_wmma_f32_16x16x32_bf16(
                  false, a, false, b, (short)0, acc, false, false);
        ap += 32;
        bp += 512;
    }
    // C/D 16x16 f32: VGPR v -> row M = v + half*8, col N = lane&15
    float bv = bias_scale * bias[coln];
    #pragma unroll
    for (int v = 0; v < 8; ++v) {
        int row = mtile * 16 + v + half * 8;
        float val = acc[v] + bv;
        if (relu) val = fmaxf(val, 0.f);
        Cf [(size_t)row * N + coln] = val;
        Cbf[(size_t)row * N + coln] = f2bf(val);
    }
}

// ======================================================================
// Output layer: [32,1024] @ Wo[1024,2] + bo   (N=2, plain VALU dots)
// ======================================================================
__global__ void out_kernel(const float* __restrict__ r2,
                           const float* __restrict__ Wo,
                           const float* __restrict__ bo,
                           float* __restrict__ out) {
    int t = threadIdx.x;                // 64 threads: (n, a)
    if (t >= NB * NA) return;
    int n = t >> 1, a = t & 1;
    float acc = bo[a];
    for (int k = 0; k < FCD; ++k)
        acc += r2[n * FCD + k] * Wo[k * NA + a];
    out[n * NA + a] = acc;
}

// ======================================================================
extern "C" void kernel_launch(void* const* d_in, const int* in_sizes, int n_in,
                              void* d_out, int out_size, void* d_ws, size_t ws_size,
                              hipStream_t stream) {
    // inputs (setup_inputs order)
    const float* image    = (const float*)d_in[0];
    const float* question = (const float*)d_in[1];
    const float* W1 = (const float*)d_in[2];
    const float* b1 = (const float*)d_in[3];
    const float* g1 = (const float*)d_in[4];
    const float* be1= (const float*)d_in[5];
    const float* W2 = (const float*)d_in[6];
    const float* b2 = (const float*)d_in[7];
    const float* g2 = (const float*)d_in[8];
    const float* be2= (const float*)d_in[9];
    const float* Wr = (const float*)d_in[10];
    const float* br = (const float*)d_in[11];
    const float* Wp = (const float*)d_in[12];
    const float* bp = (const float*)d_in[13];
    const float* Wf = (const float*)d_in[14];
    const float* bf = (const float*)d_in[15];
    const float* Wo = (const float*)d_in[16];
    const float* bo = (const float*)d_in[17];
    float* out = (float*)d_out;

    // ---- workspace layout (256B-aligned chunks) ----
    char* wsb = (char*)d_ws;
    size_t off = 0;
    auto alloc = [&](size_t bytes) -> void* {
        void* p = wsb + off;
        off = (off + bytes + 255) & ~(size_t)255;
        return p;
    };
    float* y1   = (float*)alloc((size_t)NB * C1 * H1 * H1 * 4);
    float* y2   = (float*)alloc((size_t)NB * C1 * H2 * H2 * 4);
    float* mu1  = (float*)alloc(C1 * 4);
    float* rs1  = (float*)alloc(C1 * 4);
    float* mu2  = (float*)alloc(C1 * 4);
    float* rs2  = (float*)alloc(C1 * 4);
    __bf16* ArelB = (__bf16*)alloc((size_t)NB * KRELP * 2);
    __bf16* WrP   = (__bf16*)alloc((size_t)(MREL/16) * 6  * 512 * 2);
    __bf16* WpP   = (__bf16*)alloc((size_t)(PROJ/16) * 4  * 512 * 2);
    __bf16* WfP   = (__bf16*)alloc((size_t)(FCD /16) * 16 * 512 * 2);
    float*  rel   = (float*) alloc((size_t)NB * MREL * 4);
    __bf16* relB  = (__bf16*)alloc((size_t)NB * MREL * 2);
    float*  r1    = (float*) alloc((size_t)NB * PROJ * 4);
    __bf16* r1B   = (__bf16*)alloc((size_t)NB * PROJ * 2);
    float*  r2    = (float*) alloc((size_t)NB * FCD * 4);
    __bf16* r2B   = (__bf16*)alloc((size_t)NB * FCD * 2);

    // ---- weight packing (one pass, off the critical GEMM path) ----
    {
        int tot = (MREL/16) * 6 * 512;
        pack_B_kernel<<<(tot + 255)/256, 256, 0, stream>>>(Wr, WrP, MREL, DREL, 6);
        tot = (PROJ/16) * 4 * 512;
        pack_B_kernel<<<(tot + 255)/256, 256, 0, stream>>>(Wp, WpP, PROJ, MREL, 4);
        tot = (FCD/16) * 16 * 512;
        pack_B_kernel<<<(tot + 255)/256, 256, 0, stream>>>(Wf, WfP, FCD, PROJ, 16);
    }

    // ---- stem ----
    conv1_kernel<<<(NB * C1 * H1 * H1 + 255) / 256, 256, 0, stream>>>(image, W1, b1, y1);
    bn_stats_kernel<<<C1, 256, 0, stream>>>(y1, mu1, rs1, H1 * H1);
    conv2_kernel<<<(NB * C1 * H2 * H2 + 255) / 256, 256, 0, stream>>>(
        y1, mu1, rs1, g1, be1, W2, b2, y2);
    bn_stats_kernel<<<C1, 256, 0, stream>>>(y2, mu2, rs2, H2 * H2);

    // ---- factorized pair-sum -> bf16 relation A matrix ----
    build_relA_kernel<<<NB, 256, 0, stream>>>(y2, mu2, rs2, g2, be2, question, ArelB);

    // ---- WMMA GEMM chain (bf16 in, f32 accumulate) ----
    // relations = Arel @ Wr + P*br   [32x180(->192)]@[180x128]
    gemm_bf16_wmma_kernel<<<dim3(MREL / 16, NB / 16), 32, 0, stream>>>(
        ArelB, WrP, br, rel, relB, MREL, 6, PAIRS, 0);
    // r1 = relu(rel @ Wp + bp)       [32x128]@[128x512]
    gemm_bf16_wmma_kernel<<<dim3(PROJ / 16, NB / 16), 32, 0, stream>>>(
        relB, WpP, bp, r1, r1B, PROJ, 4, 1.0f, 1);
    // r2 = relu(r1 @ Wf + bf)        [32x512]@[512x1024]
    gemm_bf16_wmma_kernel<<<dim3(FCD / 16, NB / 16), 32, 0, stream>>>(
        r1B, WfP, bf, r2, r2B, FCD, 16, 1.0f, 1);
    // out = r2 @ Wo + bo             [32x1024]@[1024x2]
    out_kernel<<<1, 64, 0, stream>>>(r2, Wo, bo, out);
}
